// S4Layer_51591147159554
// MI455X (gfx1250) — compile-verified, matrix operand order checked by reading, not run on hardware
//
#include <hip/hip_runtime.h>
#include <hip/hip_bf16.h>
#include <math.h>

typedef __attribute__((ext_vector_type(16))) __bf16 v16bf;
typedef __attribute__((ext_vector_type(8)))  __bf16 v8bf;
typedef __attribute__((ext_vector_type(8)))  float  v8f;

// Unaligned (2-byte) 8 x bf16 load wrapper: lane-shifted Toeplitz rows are not
// 16B aligned; packed struct removes the alignment assumption without any
// exec-mask predication (lowers to ds_load_b128 with byte offsets).
struct __attribute__((packed)) v8bf_pk { v8bf v; };

#define DM 512
#define NN 64
#define LL 2048
#define BB 16
#define KPAD 64
#define RG 4          // row-groups (16 rows each) per wave => 64-row wave tile

// ---------------------------------------------------------------------------
// Kernel 1: fused Cauchy kernel -> at_roots[d, l] (complex float2)
// g[d,l] = -i * (2/step_d) * tan(pi*l/L)   (pure imaginary)
// c[l]   = 1 - i*tan(pi*l/L)
// Conjugate symmetry: at_roots[d, L-l] = conj(at_roots[d, l]) (K is real),
// so compute only l in [0, L/2] and mirror.
// ---------------------------------------------------------------------------
__global__ __launch_bounds__(256) void s4_cauchy(
    const float* __restrict__ p_ri, const float* __restrict__ q_ri,
    const float* __restrict__ lam_ri, const float* __restrict__ B_ri,
    const float* __restrict__ Ct_ri, const float* __restrict__ log_step,
    float2* __restrict__ at)
{
  __shared__ float lre[NN], lim[NN];
  __shared__ float w0r[NN], w0i[NN], w1r[NN], w1i[NN];
  __shared__ float w2r[NN], w2i[NN], w3r[NN], w3i[NN];
  const int d = blockIdx.x;
  const int tid = threadIdx.x;
  if (tid < NN) {
    float2 P  = ((const float2*)p_ri)[tid];
    float2 Q  = ((const float2*)q_ri)[tid];
    float2 L  = ((const float2*)lam_ri)[tid];
    float2 Bv = ((const float2*)B_ri)[(size_t)d*NN + tid];
    float2 Cv = ((const float2*)Ct_ri)[(size_t)d*NN + tid];
    lre[tid] = L.x; lim[tid] = L.y;
    // a0 = conj(Ct_d), a1 = conj(q): w = a*b elementwise
    w0r[tid] = Cv.x*Bv.x + Cv.y*Bv.y;  w0i[tid] = Cv.x*Bv.y - Cv.y*Bv.x;
    w1r[tid] = Cv.x*P.x  + Cv.y*P.y;   w1i[tid] = Cv.x*P.y  - Cv.y*P.x;
    w2r[tid] = Q.x*Bv.x  + Q.y*Bv.y;   w2i[tid] = Q.x*Bv.y  - Q.y*Bv.x;
    w3r[tid] = Q.x*P.x   + Q.y*P.y;    w3i[tid] = Q.x*P.y   - Q.y*P.x;
  }
  __syncthreads();
  const int l = blockIdx.y * 256 + tid;
  if (l > LL/2) return;

  const float tn  = __tanf(3.14159265358979f * (float)l / (float)LL);
  const float gim = -2.0f * __expf(-log_step[d]) * tn;

  float k00r=0.f,k00i=0.f,k01r=0.f,k01i=0.f,k10r=0.f,k10i=0.f,k11r=0.f,k11i=0.f;
  #pragma unroll 8
  for (int n = 0; n < NN; ++n) {
    float ar2 = -lre[n];
    float ai2 = gim - lim[n];
    float den = __frcp_rn(ar2*ar2 + ai2*ai2);
    float rr  = ar2 * den;
    float ri  = -ai2 * den;
    k00r += w0r[n]*rr - w0i[n]*ri;  k00i += w0r[n]*ri + w0i[n]*rr;
    k01r += w1r[n]*rr - w1i[n]*ri;  k01i += w1r[n]*ri + w1i[n]*rr;
    k10r += w2r[n]*rr - w2i[n]*ri;  k10i += w2r[n]*ri + w2i[n]*rr;
    k11r += w3r[n]*rr - w3i[n]*ri;  k11i += w3r[n]*ri + w3i[n]*rr;
  }
  // at = c * (k00 - k01*k10 / (1 + k11)),  c = (1, -tn)
  float t2r = k01r*k10r - k01i*k10i;
  float t2i = k01r*k10i + k01i*k10r;
  float dr  = 1.0f + k11r, di = k11i;
  float dd  = __frcp_rn(dr*dr + di*di);
  float qr2 = (t2r*dr + t2i*di) * dd;
  float qi2 = (t2i*dr - t2r*di) * dd;
  float sr  = k00r - qr2, si = k00i - qi2;
  float arO = sr + tn * si;
  float aiO = si - tn * sr;
  at[(size_t)d*LL + l] = make_float2(arO, aiO);
  if (l > 0 && l < LL/2)
    at[(size_t)d*LL + (LL - l)] = make_float2(arO, -aiO);
}

// ---------------------------------------------------------------------------
// Kernel 2: K[d, l] = Re( FFT_2048(at_roots[d]) )[l] / 2048   (bf16 out)
// (ifft + [0, L-1..1] reorder == forward FFT / L)
// 2048-pt radix-2 Stockham FFT in LDS, one workgroup per channel.
// ---------------------------------------------------------------------------
__global__ __launch_bounds__(256) void s4_fftK(
    const float2* __restrict__ at, __bf16* __restrict__ Kf)
{
  __shared__ float2 bufA[LL], bufB[LL];
  const int d = blockIdx.x, tid = threadIdx.x;
  for (int i = tid; i < LL; i += 256) bufA[i] = at[(size_t)d*LL + i];
  __syncthreads();
  float2* src = bufA; float2* dst = bufB;
  int n = LL, s = 1, ls = 0;
  while (n > 1) {
    const int half = n >> 1;
    for (int t = tid; t < LL/2; t += 256) {
      const int q = t & (s - 1);
      const int p = t >> ls;
      float2 a = src[q + s*p];
      float2 b = src[q + s*(p + half)];
      float ang = -6.28318530717959f * (float)p / (float)n;
      float sn, cs;
      __sincosf(ang, &sn, &cs);
      float2 e; e.x = a.x + b.x; e.y = a.y + b.y;
      float ox = a.x - b.x, oy = a.y - b.y;
      float2 o; o.x = ox*cs - oy*sn; o.y = ox*sn + oy*cs;
      dst[q + s*(2*p)]     = e;
      dst[q + s*(2*p + 1)] = o;
    }
    __syncthreads();
    float2* tmp = src; src = dst; dst = tmp;
    n = half; s <<= 1; ++ls;
  }
  for (int i = tid; i < LL; i += 256)
    Kf[(size_t)d*LL + i] = (__bf16)(src[i].x * (1.0f / (float)LL));
}

// ---------------------------------------------------------------------------
// Kernel 3: transpose+downconvert u[b, l, d] f32 -> Ut[d, b, l] bf16
// so WMMA B-fragments become contiguous 16B loads per lane.
// ---------------------------------------------------------------------------
__global__ __launch_bounds__(256) void s4_transpose(
    const float* __restrict__ u, __bf16* __restrict__ Ut)
{
  __shared__ float tile[32][33];
  const int b  = blockIdx.z;
  const int l0 = blockIdx.y * 32;
  const int d0 = blockIdx.x * 32;
  const int tx = threadIdx.x & 31, ty = threadIdx.x >> 5;
  for (int i = ty; i < 32; i += 8)
    tile[i][tx] = u[((size_t)b*LL + (l0 + i))*DM + d0 + tx];
  __syncthreads();
  for (int i = ty; i < 32; i += 8)
    Ut[(((size_t)(d0 + i))*BB + b)*LL + l0 + tx] = (__bf16)tile[tx][i];
}

// ---------------------------------------------------------------------------
// Kernel 4: causal Toeplitz GEMM per channel with WMMA bf16 -> f32.
//   y[b, m, d] = sum_{k<=m} K[d, m-k] * u[b, k, d] + D[d]*u[b, m, d]
// Per channel d: C(2048x16) = T(K_d)(2048x2048, lower-tri Toeplitz) * Ut_d(2048x16).
//
// A-operand: LDS holds K *reversed* with a zero tail:
//   sR[j] = K[2047 - j]  (j < 2048),  sR[2048..2111] = 0
// so fragment groups are contiguous ds_load_b128s and the causal zero mask
// comes from the tail — no compares, no branches. Adjacent 16-row groups
// share the sliding window (offset -16 halfs), so RG=4 row groups cost only
// ~5 merged ds_load_b128s, loaded in-iteration (LDS latency is short).
//
// Each wave owns a 64-row tile (RG=4 accumulators): each B panel
// (2 global_load_b128s) feeds FOUR wmmas. Only the long-latency global B
// panel is software-pipelined (2-deep), and the loop is unrolled 2x so the
// bl/bh rotation alternates registers and emits no moves.
// fp32 accumulate, fused D*u epilogue.
// ---------------------------------------------------------------------------
__global__ __launch_bounds__(256) void s4_toeplitz_wmma(
    const __bf16* __restrict__ Kf, const __bf16* __restrict__ Ut,
    const float* __restrict__ u, const float* __restrict__ Dv,
    float* __restrict__ y)
{
  __shared__ __bf16 sR[LL + KPAD];
  const int d     = blockIdx.x;
  const int m0blk = blockIdx.y * 512;       // 8 waves x 64 rows
  const int tid   = threadIdx.x;
  const int wave  = tid >> 5, lane = tid & 31;

  // reversed K fill (coalesced global reads) + zero tail
  for (int i = tid; i < LL; i += 256)
    sR[LL - 1 - i] = Kf[(size_t)d*LL + i];
  if (tid < KPAD) sR[LL + tid] = (__bf16)0.0f;
  __syncthreads();

  const int m0   = m0blk + wave * (16 * RG);   // wave tile: rows [m0, m0+64)
  const int mrow = m0 + (lane & 15);
  const int kofs = (lane & 16) ? 8 : 0;        // 16-bit operand lane->k map
  const int ncol = lane & 15;
  const __bf16* bbase = Ut + ((size_t)d*BB + ncol)*LL + kofs;
  const __bf16* abase = sR + (LL - 1) - mrow + kofs;  // row group 0

  v8f acc[RG];
  #pragma unroll
  for (int r = 0; r < RG; ++r) acc[r] = (v8f){};

  const int kiters = (m0 + 16*RG) >> 5;        // causal; m0+64 multiple of 32

  // ---- B-panel pipeline prologue (k0 = 0) ----
  v8bf bl0 = *(const v8bf*)(bbase);
  v8bf bh0 = *(const v8bf*)(bbase + 16);

  int k0 = 0;
  #pragma unroll 2
  for (int it = 1; it < kiters; ++it) {
    // issue NEXT B panel's global loads first (overlap with wmmas below)
    __builtin_prefetch(bbase + k0 + 160, 0, 3);   // global_prefetch_b8
    v8bf bl1 = *(const v8bf*)(bbase + k0 + 32);
    v8bf bh1 = *(const v8bf*)(bbase + k0 + 48);

    // A fragments for CURRENT k0 from LDS (merged ds_load_b128 window)
    v8bf xl[RG], xh[RG];
    #pragma unroll
    for (int r = 0; r < RG; ++r) {
      xl[r] = ((const v8bf_pk*)(abase - 16*r + k0))->v;
      xh[r] = ((const v8bf_pk*)(abase - 16*r + k0 + 16))->v;
    }

    v16bf b = __builtin_shufflevector(bl0, bh0,
        0,1,2,3,4,5,6,7,8,9,10,11,12,13,14,15);
    #pragma unroll
    for (int r = 0; r < RG; ++r) {
      v16bf a = __builtin_shufflevector(xl[r], xh[r],
          0,1,2,3,4,5,6,7,8,9,10,11,12,13,14,15);
      acc[r] = __builtin_amdgcn_wmma_f32_16x16x32_bf16(
          false, a, false, b, (short)0, acc[r], false, false);
    }

    bl0 = bl1; bh0 = bh1;                     // vanishes under unroll-2
    k0 += 32;
  }

  // ---- pipeline drain: last B panel ----
  {
    v8bf xl[RG], xh[RG];
    #pragma unroll
    for (int r = 0; r < RG; ++r) {
      xl[r] = ((const v8bf_pk*)(abase - 16*r + k0))->v;
      xh[r] = ((const v8bf_pk*)(abase - 16*r + k0 + 16))->v;
    }
    v16bf b = __builtin_shufflevector(bl0, bh0,
        0,1,2,3,4,5,6,7,8,9,10,11,12,13,14,15);
    #pragma unroll
    for (int r = 0; r < RG; ++r) {
      v16bf a = __builtin_shufflevector(xl[r], xh[r],
          0,1,2,3,4,5,6,7,8,9,10,11,12,13,14,15);
      acc[r] = __builtin_amdgcn_wmma_f32_16x16x32_bf16(
          false, a, false, b, (short)0, acc[r], false, false);
    }
  }

  // C/D layout: VGPR j -> M = base + j (+8 for lanes 16-31), N = lane&15
  const float Dd = Dv[d];
  const int mb = m0 + ((lane >> 4) << 3);
  #pragma unroll
  for (int r = 0; r < RG; ++r) {
    #pragma unroll
    for (int j = 0; j < 8; ++j) {
      const int m = mb + 16*r + j;
      const size_t idx = ((size_t)ncol*LL + m)*DM + d;
      y[idx] = acc[r][j] + Dd * u[idx];
    }
  }
}

// ---------------------------------------------------------------------------
extern "C" void kernel_launch(void* const* d_in, const int* in_sizes, int n_in,
                              void* d_out, int out_size, void* d_ws, size_t ws_size,
                              hipStream_t stream) {
  const float* u        = (const float*)d_in[0];
  const float* p_ri     = (const float*)d_in[1];
  const float* q_ri     = (const float*)d_in[2];
  const float* lam_ri   = (const float*)d_in[3];
  const float* B_ri     = (const float*)d_in[4];
  const float* Ct_ri    = (const float*)d_in[5];
  const float* Dv       = (const float*)d_in[6];
  const float* log_step = (const float*)d_in[7];
  float* y = (float*)d_out;

  char* ws = (char*)d_ws;
  const size_t off_at = 0;                                      // 8 MB float2
  const size_t off_kf = off_at + (size_t)DM*LL*sizeof(float2);  // 2 MB bf16
  const size_t off_ut = off_kf + (size_t)DM*LL*sizeof(__bf16);  // 32 MB bf16
  float2* at = (float2*)(ws + off_at);
  __bf16* Kf = (__bf16*)(ws + off_kf);
  __bf16* Ut = (__bf16*)(ws + off_ut);

  s4_cauchy<<<dim3(DM, 5), 256, 0, stream>>>(p_ri, q_ri, lam_ri, B_ri, Ct_ri,
                                             log_step, at);
  s4_fftK<<<dim3(DM), 256, 0, stream>>>(at, Kf);
  s4_transpose<<<dim3(DM/32, LL/32, BB), 256, 0, stream>>>(u, Ut);
  // 8 waves x 64 rows = 512 rows per block
  s4_toeplitz_wmma<<<dim3(DM, LL/512), 256, 0, stream>>>(Kf, Ut, u, Dv, y);
}
